// Model_24378234372406
// MI455X (gfx1250) — matrix operand order repaired and checked by reading, not compiled, not run on hardware
//
#include <hip/hip_runtime.h>
#include <math.h>

// ---------------------------------------------------------------------------
// Model constants
// ---------------------------------------------------------------------------
#define BATCH_   32
#define CVAR_    21
#define SEQ_     672
#define PRED_    336
#define D_MODEL  512
#define DFF_     2048
#define N_TOK    84            // tokens per variate (R*T = 14*6)
#define N_REG    14
#define T_PER_R  6
#define N_BC     (BATCH_*CVAR_)        // 672
#define M_TOK    (N_BC*N_TOK)          // 56448 rows for token GEMMs
#define K_HEAD   (D_MODEL*N_TOK)       // 43008

typedef _Float16 v16h __attribute__((ext_vector_type(16)));
typedef _Float16 h8   __attribute__((ext_vector_type(8)));
typedef float    v8f  __attribute__((ext_vector_type(8)));
typedef int      v4i  __attribute__((vector_size(16)));

union FragH { v16h v; h8 h[2]; _Float16 e[16]; };

static __device__ __forceinline__ v8f wmma_f16(v16h a, v16h b, v8f c) {
  // D = A(16x32 f16) x B(32x16 f16) + C(16x16 f32)
  return __builtin_amdgcn_wmma_f32_16x16x32_f16(false, a, false, b, (short)0, c,
                                                false, false);
}

// ----- gfx1250 async global->LDS staging (guarded; falls back if absent) ----
#if defined(__has_builtin)
#if __has_builtin(__builtin_amdgcn_global_load_async_to_lds_b128) && \
    __has_builtin(__builtin_amdgcn_s_wait_asynccnt)
#define USE_ASYNC_LDS 1
#endif
#endif
#ifndef USE_ASYNC_LDS
#define USE_ASYNC_LDS 0
#endif

// builtin expects (v4i addrspace(1)* gsrc, v4i addrspace(3)* ldst, imm, imm)
#define AS1P(p) ((__attribute__((address_space(1))) v4i*)((void*)(p)))
#define AS3P(p) ((__attribute__((address_space(3))) v4i*)((void*)(p)))

// ---------------------------------------------------------------------------
// Generic f32 -> f16 cast (weights)
// ---------------------------------------------------------------------------
__global__ void cast_f32_f16_kernel(const float* __restrict__ s,
                                    _Float16* __restrict__ d, long long n) {
  for (long long i = (long long)blockIdx.x * blockDim.x + threadIdx.x; i < n;
       i += (long long)gridDim.x * blockDim.x)
    d[i] = (_Float16)s[i];
}

// ---------------------------------------------------------------------------
// Prep: instance-norm over time, router MLP + argmax, patch embedding + PE.
// One workgroup per (b,c) series.  Writes Z (f32 residual stream), ZH (f16).
// ---------------------------------------------------------------------------
__global__ void __launch_bounds__(256)
prep_kernel(const float* __restrict__ x_enc,
            const float* __restrict__ cls_w1, const float* __restrict__ cls_b1,
            const float* __restrict__ cls_w2, const float* __restrict__ cls_b2,
            const float* __restrict__ e0, const float* __restrict__ e1,
            const float* __restrict__ e2, const float* __restrict__ e3,
            float* __restrict__ Z, _Float16* __restrict__ ZH,
            float* __restrict__ MEAN, float* __restrict__ STDV) {
  const int bc = blockIdx.x;
  const int b  = bc / CVAR_;
  const int c  = bc - b * CVAR_;
  const int tid = threadIdx.x;

  __shared__ float xn[SEQ_];
  __shared__ float red[256];
  __shared__ int   pred[N_REG];

  float s = 0.f;
  for (int t = tid; t < SEQ_; t += 256) {
    float v = x_enc[((long long)b * SEQ_ + t) * CVAR_ + c];
    xn[t] = v;
    s += v;
  }
  red[tid] = s; __syncthreads();
  for (int o = 128; o > 0; o >>= 1) { if (tid < o) red[tid] += red[tid + o]; __syncthreads(); }
  const float mean = red[0] * (1.0f / SEQ_);
  __syncthreads();

  float s2 = 0.f;
  for (int t = tid; t < SEQ_; t += 256) { float d = xn[t] - mean; s2 += d * d; }
  red[tid] = s2; __syncthreads();
  for (int o = 128; o > 0; o >>= 1) { if (tid < o) red[tid] += red[tid + o]; __syncthreads(); }
  const float stdev = sqrtf(red[0] * (1.0f / SEQ_) + 1e-5f);
  if (tid == 0) { MEAN[bc] = mean; STDV[bc] = stdev; }
  __syncthreads();

  const float inv = 1.0f / stdev;
  for (int t = tid; t < SEQ_; t += 256) xn[t] = (xn[t] - mean) * inv;
  __syncthreads();

  // Router: Linear(48,64)+ReLU -> Linear(64,4) -> argmax, one thread per region
  if (tid < N_REG) {
    const float* xr = &xn[tid * 48];
    float hdn[64];
    for (int j = 0; j < 64; ++j) {
      float a = cls_b1[j];
      for (int p = 0; p < 48; ++p) a += xr[p] * cls_w1[j * 48 + p];
      hdn[j] = a > 0.f ? a : 0.f;
    }
    float best = -3.0e38f; int bestE = 0;
    for (int e = 0; e < 4; ++e) {
      float a = cls_b2[e];
      for (int j = 0; j < 64; ++j) a += hdn[j] * cls_w2[e * 64 + j];
      if (a > best) { best = a; bestE = e; }
    }
    pred[tid] = bestE;
  }
  __syncthreads();

  const float* embs[4] = { e0, e1, e2, e3 };
  const int plens[4] = { 8, 16, 24, 48 };
  for (int o = tid; o < N_TOK * D_MODEL; o += 256) {
    const int n = o >> 9;            // token
    const int d = o & 511;           // model dim
    const int r  = n / T_PER_R;
    const int tt = n - r * T_PER_R;
    const int e  = pred[r];
    const int p  = plens[e];
    const int num = 48 / p;
    const int rep = T_PER_R - num + 1;
    int idx = tt / rep; if (idx > num - 1) idx = num - 1;
    const float* patch = &xn[r * 48 + idx * p];
    const float* w = embs[e] + (long long)d * p;
    float acc = 0.f;
    for (int pp = 0; pp < p; ++pp) acc += patch[pp] * w[pp];
    // sinusoidal PE
    const int j2 = d >> 1;
    const float div = expf((float)(2 * j2) * (-9.210340371976184f / (float)D_MODEL));
    const float ang = (float)n * div;
    const float pe = (d & 1) ? cosf(ang) : sinf(ang);
    const float val = acc + pe;
    const long long gi = ((long long)bc * N_TOK + n) * D_MODEL + d;
    Z[gi]  = val;
    ZH[gi] = (_Float16)val;
  }
}

// ---------------------------------------------------------------------------
// WMMA GEMM:  out[M,N] = act(A[M,K](f16) * W[N,K](f16)^T + bias[N])
// Block tile 128x128, K-step 32, 128 threads = 4 waves (2x2), each wave owns
// a 64x64 sub-tile = 4x4 WMMA 16x16 tiles -> 16 WMMAs per 16 ds_load_b128.
// Full tiles use gfx1250 async global->LDS double-buffered pipeline.
// ---------------------------------------------------------------------------
#define BM 128
#define BN 128
#define BK 32
#define LDT 40   // padded LDS row stride (halves); 80B rows keep 16B alignment

static __device__ __forceinline__
void gemm_stage_manual(const _Float16* __restrict__ A, const _Float16* __restrict__ W,
                       _Float16* dA, _Float16* dB, int m0, int n0, int kb,
                       int M, int N, int K, int KT, int kt, int tid) {
  for (int ch = tid; ch < 512; ch += 128) {
    const int r  = ch >> 2;
    const int c8 = (ch & 3) << 3;
    h8 va = {0,0,0,0,0,0,0,0};
    h8 vb = {0,0,0,0,0,0,0,0};
    const int ga = m0 + r;
    if (ga < M) {
      va = *(const h8*)(A + (long long)ga * K + kb + c8);
      if (kt + 1 < KT)
        __builtin_prefetch(A + (long long)ga * K + kb + BK + c8, 0, 1);
    }
    const int gb = n0 + r;
    if (gb < N) {
      vb = *(const h8*)(W + (long long)gb * K + kb + c8);
      if (kt + 1 < KT)
        __builtin_prefetch(W + (long long)gb * K + kb + BK + c8, 0, 1);
    }
    *(h8*)(dA + r * LDT + c8) = va;
    *(h8*)(dB + r * LDT + c8) = vb;
  }
}

#if USE_ASYNC_LDS
static __device__ __forceinline__
void gemm_stage_async(const _Float16* A, const _Float16* W,
                      _Float16* dA, _Float16* dB, int m0, int n0, int kb,
                      int K, int tid) {
  for (int ch = tid; ch < 512; ch += 128) {
    const int r  = ch >> 2;
    const int c8 = (ch & 3) << 3;
    __builtin_amdgcn_global_load_async_to_lds_b128(
        AS1P(A + (long long)(m0 + r) * K + kb + c8),
        AS3P(dA + r * LDT + c8), 0, 0);
    __builtin_amdgcn_global_load_async_to_lds_b128(
        AS1P(W + (long long)(n0 + r) * K + kb + c8),
        AS3P(dB + r * LDT + c8), 0, 0);
  }
}
#endif

static __device__ __forceinline__
void gemm_compute(const _Float16* dA, const _Float16* dB, int lane, int wm, int wn,
                  v8f acc[4][4]) {
  FragH af[4], bf[4];
  // A frag: lanes 0-15 hold K 0..15, lanes 16-31 K 16..31
  const int arow = wm * 64 + (lane & 15);
  const int akb  = (lane >> 4) << 4;
  for (int mt = 0; mt < 4; ++mt) {
    const _Float16* p = dA + (arow + mt * 16) * LDT + akb;
    af[mt].h[0] = *(const h8*)(p);
    af[mt].h[1] = *(const h8*)(p + 8);
  }
  // B frag: lane n holds W-row (output col) K 0..15 / 16..31
  const int bkb = (lane >> 4) << 4;
  for (int nt = 0; nt < 4; ++nt) {
    const _Float16* p = dB + (wn * 64 + nt * 16 + (lane & 15)) * LDT + bkb;
    bf[nt].h[0] = *(const h8*)(p);
    bf[nt].h[1] = *(const h8*)(p + 8);
  }
  for (int mt = 0; mt < 4; ++mt)
    for (int nt = 0; nt < 4; ++nt)
      acc[mt][nt] = wmma_f16(af[mt].v, bf[nt].v, acc[mt][nt]);
}

template <bool OUT_HALF, bool DO_GELU>
__global__ void __launch_bounds__(128)
wmma_gemm_kernel(const _Float16* __restrict__ A, const _Float16* __restrict__ W,
                 const float* __restrict__ bias, void* __restrict__ outp,
                 int M, int N, int K) {
  __shared__ _Float16 sA[2 * BM * LDT];
  __shared__ _Float16 sB[2 * BM * LDT];

  const int tid  = threadIdx.x;
  const int lane = tid & 31;
  const int wave = tid >> 5;            // 0..3
  const int wm   = wave >> 1;           // 0..1 -> 64-row strip
  const int wn   = wave & 1;            // 0..1 -> 64-col strip
  const int m0   = blockIdx.y * BM;
  const int n0   = blockIdx.x * BN;
  const int KT   = K / BK;

  const v8f vz = {0.f,0.f,0.f,0.f,0.f,0.f,0.f,0.f};
  v8f acc[4][4];
  for (int i = 0; i < 4; ++i) for (int j = 0; j < 4; ++j) acc[i][j] = vz;

#if USE_ASYNC_LDS
  const bool full = (m0 + BM <= M) && (n0 + BN <= N);
  if (full) {
    // double-buffered async pipeline: stage k+1 while computing k
    gemm_stage_async(A, W, sA, sB, m0, n0, 0, K, tid);
    for (int kt = 0; kt < KT; ++kt) {
      __builtin_amdgcn_s_wait_asynccnt(0);
      __syncthreads();
      const int buf = kt & 1;
      if (kt + 1 < KT)
        gemm_stage_async(A, W, sA + (buf ^ 1) * (BM * LDT), sB + (buf ^ 1) * (BM * LDT),
                         m0, n0, (kt + 1) * BK, K, tid);
      gemm_compute(sA + buf * (BM * LDT), sB + buf * (BM * LDT), lane, wm, wn, acc);
      __syncthreads();
    }
  } else
#endif
  {
    for (int kt = 0; kt < KT; ++kt) {
      const int buf = kt & 1;
      gemm_stage_manual(A, W, sA + buf * (BM * LDT), sB + buf * (BM * LDT),
                        m0, n0, kt * BK, M, N, K, KT, kt, tid);
      __syncthreads();
      gemm_compute(sA + buf * (BM * LDT), sB + buf * (BM * LDT), lane, wm, wn, acc);
      __syncthreads();
    }
  }

  // epilogue: C/D layout -> row = strip + 8*(lane>=16) + vgpr, col = strip + (lane&15)
  for (int mt = 0; mt < 4; ++mt)
    for (int nt = 0; nt < 4; ++nt)
      for (int el = 0; el < 8; ++el) {
        const int gr = m0 + wm * 64 + mt * 16 + ((lane >> 4) << 3) + el;
        const int gc = n0 + wn * 64 + nt * 16 + (lane & 15);
        if (gr < M && gc < N) {
          float x = acc[mt][nt][el] + bias[gc];
          if (DO_GELU) x = 0.5f * x * (1.0f + erff(x * 0.7071067811865475f));
          if (OUT_HALF) ((_Float16*)outp)[(long long)gr * N + gc] = (_Float16)x;
          else          ((float*)outp)[(long long)gr * N + gc] = x;
        }
      }
}

// ---------------------------------------------------------------------------
// Fused attention for one (bc, head): S = QK^T/8, softmax, O = P V.
// 84 tokens padded to 96; 6 waves, each owns one 16-row strip.  All WMMA.
// LDS: Q,K,V (96x64 f16 each) + P (96x96 f16); S(f16) overlays Q/K region.
// ---------------------------------------------------------------------------
__global__ void __launch_bounds__(192)
attention_kernel(const _Float16* __restrict__ Q, const _Float16* __restrict__ Km,
                 const _Float16* __restrict__ V, _Float16* __restrict__ O) {
  const int bc   = blockIdx.x;
  const int h    = blockIdx.y;
  const int tid  = threadIdx.x;
  const int lane = tid & 31;
  const int wave = tid >> 5;   // 0..5

  __shared__ _Float16 smem[3 * 96 * 64 + 96 * 96];
  _Float16* sQ = smem;
  _Float16* sK = smem + 96 * 64;
  _Float16* sV = smem + 2 * 96 * 64;
  _Float16* sP = smem + 3 * 96 * 64;
  _Float16* sS = smem;  // overlays sQ/sK after they are consumed (9216 halves)

  const long long base = ((long long)bc * N_TOK) * D_MODEL + h * 64;

  for (int ch = tid; ch < 768; ch += 192) {       // 96 rows * 8 chunks
    const int sidx = ch >> 3;
    const int c8   = (ch & 7) << 3;
    h8 zq = {0,0,0,0,0,0,0,0}, zk = zq, zv = zq;
    if (sidx < N_TOK) {
      const long long gi = base + (long long)sidx * D_MODEL + c8;
      zq = *(const h8*)(Q + gi);
      zk = *(const h8*)(Km + gi);
      zv = *(const h8*)(V + gi);
    }
    *(h8*)(sQ + sidx * 64 + c8) = zq;
    *(h8*)(sK + sidx * 64 + c8) = zk;
    *(h8*)(sV + sidx * 64 + c8) = zv;
  }
  __syncthreads();

  // ---- S = Q K^T (scaled), kept in registers through the barrier ----
  const v8f vz = {0.f,0.f,0.f,0.f,0.f,0.f,0.f,0.f};
  v8f accs[6];
  for (int i = 0; i < 6; ++i) accs[i] = vz;
  for (int kk = 0; kk < 64; kk += 32) {
    FragH a;
    const int arow = wave * 16 + (lane & 15);
    const int akb  = kk + ((lane >> 4) << 4);
    a.h[0] = *(const h8*)(sQ + arow * 64 + akb);
    a.h[1] = *(const h8*)(sQ + arow * 64 + akb + 8);
    const int bkb = kk + ((lane >> 4) << 4);
    for (int nt = 0; nt < 6; ++nt) {
      FragH bfr;
      const _Float16* p = sK + (nt * 16 + (lane & 15)) * 64 + bkb;
      bfr.h[0] = *(const h8*)(p);
      bfr.h[1] = *(const h8*)(p + 8);
      accs[nt] = wmma_f16(a.v, bfr.v, accs[nt]);
    }
  }
  __syncthreads();  // everyone done reading sQ/sK before S overlays them

  for (int nt = 0; nt < 6; ++nt)
    for (int el = 0; el < 8; ++el) {
      const int r = wave * 16 + ((lane >> 4) << 3) + el;
      const int cc = nt * 16 + (lane & 15);
      sS[r * 96 + cc] = (_Float16)(accs[nt][el] * 0.125f);  // 1/sqrt(64)
    }
  __syncthreads();

  // ---- row softmax over the 84 valid columns; zero the padding ----
  if (tid < 96) {
    if (tid < N_TOK) {
      float mx = -3.0e38f;
      for (int cc = 0; cc < N_TOK; ++cc) mx = fmaxf(mx, (float)sS[tid * 96 + cc]);
      float sum = 0.f;
      for (int cc = 0; cc < N_TOK; ++cc) sum += expf((float)sS[tid * 96 + cc] - mx);
      const float inv = 1.0f / sum;
      for (int cc = 0; cc < 96; ++cc)
        sP[tid * 96 + cc] = (cc < N_TOK)
            ? (_Float16)(expf((float)sS[tid * 96 + cc] - mx) * inv)
            : (_Float16)0.f;
    } else {
      for (int cc = 0; cc < 96; ++cc) sP[tid * 96 + cc] = (_Float16)0.f;
    }
  }
  __syncthreads();

  // ---- O = P V ----
  v8f acco[4];
  for (int i = 0; i < 4; ++i) acco[i] = vz;
  for (int kk = 0; kk < 96; kk += 32) {
    FragH a;
    const int arow = wave * 16 + (lane & 15);
    const int akb  = kk + ((lane >> 4) << 4);
    a.h[0] = *(const h8*)(sP + arow * 96 + akb);
    a.h[1] = *(const h8*)(sP + arow * 96 + akb + 8);
    const int sb = kk + ((lane >> 4) << 4);
    for (int nt = 0; nt < 4; ++nt) {
      FragH bfr;
      const int dcol = nt * 16 + (lane & 15);
      for (int j = 0; j < 16; ++j) bfr.e[j] = sV[(sb + j) * 64 + dcol];
      acco[nt] = wmma_f16(a.v, bfr.v, acco[nt]);
    }
  }
  for (int nt = 0; nt < 4; ++nt)
    for (int el = 0; el < 8; ++el) {
      const int r = wave * 16 + ((lane >> 4) << 3) + el;
      const int d = nt * 16 + (lane & 15);
      if (r < N_TOK)
        O[base + (long long)r * D_MODEL + d] = (_Float16)acco[nt][el];
    }
}

// ---------------------------------------------------------------------------
// z = LayerNorm(zin + tmp) * g + b ; writes fp32 residual and f16 copy.
// One block per token row (512 cols, 256 threads x2).
// ---------------------------------------------------------------------------
__global__ void __launch_bounds__(256)
add_ln_kernel(const float* __restrict__ zin, const float* __restrict__ tmp,
              const float* __restrict__ g, const float* __restrict__ bb,
              float* __restrict__ zout, _Float16* __restrict__ zh) {
  __shared__ float red[256];
  const long long row = blockIdx.x;
  const int tid = threadIdx.x;
  const float* zr = zin + row * D_MODEL;
  const float* tr = tmp + row * D_MODEL;
  const float v0 = zr[tid] + tr[tid];
  const float v1 = zr[tid + 256] + tr[tid + 256];
  red[tid] = v0 + v1; __syncthreads();
  for (int o = 128; o > 0; o >>= 1) { if (tid < o) red[tid] += red[tid + o]; __syncthreads(); }
  const float mean = red[0] * (1.0f / D_MODEL);
  __syncthreads();
  const float d0 = v0 - mean, d1 = v1 - mean;
  red[tid] = d0 * d0 + d1 * d1; __syncthreads();
  for (int o = 128; o > 0; o >>= 1) { if (tid < o) red[tid] += red[tid + o]; __syncthreads(); }
  const float rstd = rsqrtf(red[0] * (1.0f / D_MODEL) + 1e-5f);
  const float o0 = d0 * rstd * g[tid] + bb[tid];
  const float o1 = d1 * rstd * g[tid + 256] + bb[tid + 256];
  zout[row * D_MODEL + tid] = o0;
  zout[row * D_MODEL + tid + 256] = o1;
  zh[row * D_MODEL + tid] = (_Float16)o0;
  zh[row * D_MODEL + tid + 256] = (_Float16)o1;
}

// ---------------------------------------------------------------------------
// BatchNorm(eval, unit stats) + [bc,n,d] -> [bc, d*84+n] f16 head input.
// ---------------------------------------------------------------------------
__global__ void bn_head_kernel(const float* __restrict__ Z, const float* __restrict__ g,
                               const float* __restrict__ b, _Float16* __restrict__ HIN) {
  const float inv = rsqrtf(1.0f + 1e-5f);
  const long long total = (long long)M_TOK * D_MODEL;
  for (long long i = (long long)blockIdx.x * blockDim.x + threadIdx.x; i < total;
       i += (long long)gridDim.x * blockDim.x) {
    const long long bc  = i / ((long long)N_TOK * D_MODEL);
    const long long rem = i - bc * ((long long)N_TOK * D_MODEL);
    const int n = (int)(rem >> 9);
    const int d = (int)(rem & 511);
    const float val = Z[i] * inv * g[d] + b[d];
    HIN[bc * K_HEAD + (long long)d * N_TOK + n] = (_Float16)val;
  }
}

// ---------------------------------------------------------------------------
// De-normalize: out[b,t,c] = head[bc,t]*std[bc] + mean[bc]
// ---------------------------------------------------------------------------
__global__ void finalize_kernel(const float* __restrict__ HO, const float* __restrict__ MEAN,
                                const float* __restrict__ STDV, float* __restrict__ out) {
  const int total = BATCH_ * PRED_ * CVAR_;
  for (int i = blockIdx.x * blockDim.x + threadIdx.x; i < total;
       i += gridDim.x * blockDim.x) {
    const int b   = i / (PRED_ * CVAR_);
    const int rem = i - b * (PRED_ * CVAR_);
    const int t   = rem / CVAR_;
    const int c   = rem - t * CVAR_;
    const int bc  = b * CVAR_ + c;
    out[i] = HO[(long long)bc * PRED_ + t] * STDV[bc] + MEAN[bc];
  }
}

// ---------------------------------------------------------------------------
// Host orchestration
// ---------------------------------------------------------------------------
extern "C" void kernel_launch(void* const* d_in, const int* in_sizes, int n_in,
                              void* d_out, int out_size, void* d_ws, size_t ws_size,
                              hipStream_t stream) {
  (void)in_sizes; (void)n_in; (void)out_size; (void)ws_size;
  const float* x_enc  = (const float*)d_in[0];
  const float* cls_w1 = (const float*)d_in[4];
  const float* cls_b1 = (const float*)d_in[5];
  const float* cls_w2 = (const float*)d_in[6];
  const float* cls_b2 = (const float*)d_in[7];
  const float* emb_w0 = (const float*)d_in[8];
  const float* emb_w1 = (const float*)d_in[9];
  const float* emb_w2 = (const float*)d_in[10];
  const float* emb_w3 = (const float*)d_in[11];
  const float* wq = (const float*)d_in[12];
  const float* bq = (const float*)d_in[13];
  const float* wk = (const float*)d_in[14];
  const float* bk = (const float*)d_in[15];
  const float* wv = (const float*)d_in[16];
  const float* bv = (const float*)d_in[17];
  const float* wo = (const float*)d_in[18];
  const float* bo = (const float*)d_in[19];
  const float* ln1_g = (const float*)d_in[20];
  const float* ln1_b = (const float*)d_in[21];
  const float* ln2_g = (const float*)d_in[22];
  const float* ln2_b = (const float*)d_in[23];
  const float* ff1_w = (const float*)d_in[24];
  const float* ff1_b = (const float*)d_in[25];
  const float* ff2_w = (const float*)d_in[26];
  const float* ff2_b = (const float*)d_in[27];
  const float* bn_g = (const float*)d_in[28];
  const float* bn_b = (const float*)d_in[29];
  const float* head_w = (const float*)d_in[30];
  const float* head_b = (const float*)d_in[31];
  float* out = (float*)d_out;

  // ----- workspace arena -----
  char* ws = (char*)d_ws;
  size_t off = 0;
  auto carve = [&](size_t bytes) {
    size_t o = off;
    off += (bytes + 255) & ~(size_t)255;
    return o;
  };
  float*    MEAN = (float*)(ws + carve((size_t)N_BC * 4));
  float*    STDV = (float*)(ws + carve((size_t)N_BC * 4));
  float*    Z    = (float*)(ws + carve((size_t)M_TOK * D_MODEL * 4));
  _Float16* ZH   = (_Float16*)(ws + carve((size_t)M_TOK * D_MODEL * 2));
  char*     RC   = ws + carve((size_t)M_TOK * DFF_ * 2);  // union: qkv+attn | hidden
  _Float16* Qb   = (_Float16*)(RC);
  _Float16* Kb   = (_Float16*)(RC + (size_t)M_TOK * D_MODEL * 2);
  _Float16* Vb   = (_Float16*)(RC + (size_t)M_TOK * D_MODEL * 4);
  _Float16* ATTN = (_Float16*)(RC + (size_t)M_TOK * D_MODEL * 6);
  _Float16* HIDD = (_Float16*)(RC);
  float*    TMP  = (float*)(ws + carve((size_t)M_TOK * D_MODEL * 4));
  _Float16* WQH  = (_Float16*)(ws + carve((size_t)2 * D_MODEL * D_MODEL * 2));
  _Float16* WKH  = (_Float16*)(ws + carve((size_t)2 * D_MODEL * D_MODEL * 2));
  _Float16* WVH  = (_Float16*)(ws + carve((size_t)2 * D_MODEL * D_MODEL * 2));
  _Float16* WOH  = (_Float16*)(ws + carve((size_t)2 * D_MODEL * D_MODEL * 2));
  _Float16* FF1H = (_Float16*)(ws + carve((size_t)2 * DFF_ * D_MODEL * 2));
  _Float16* FF2H = (_Float16*)(ws + carve((size_t)2 * D_MODEL * DFF_ * 2));
  _Float16* HWH  = (_Float16*)(ws + carve((size_t)PRED_ * K_HEAD * 2));
  float*    HOUT = (float*)(ws + carve((size_t)N_BC * PRED_ * 4));
  _Float16* HIN  = ZH;  // reuse: ZH dead after final LN (head reads f16 transpose)

  auto castN = [&](const float* s, _Float16* d, long long n) {
    int blocks = (int)((n + 255) / 256);
    if (blocks > 65535) blocks = 65535;
    cast_f32_f16_kernel<<<blocks, 256, 0, stream>>>(s, d, n);
  };
  castN(wq, WQH, 2LL * D_MODEL * D_MODEL);
  castN(wk, WKH, 2LL * D_MODEL * D_MODEL);
  castN(wv, WVH, 2LL * D_MODEL * D_MODEL);
  castN(wo, WOH, 2LL * D_MODEL * D_MODEL);
  castN(ff1_w, FF1H, 2LL * DFF_ * D_MODEL);
  castN(ff2_w, FF2H, 2LL * D_MODEL * DFF_);
  castN(head_w, HWH, (long long)PRED_ * K_HEAD);

  prep_kernel<<<N_BC, 256, 0, stream>>>(x_enc, cls_w1, cls_b1, cls_w2, cls_b2,
                                        emb_w0, emb_w1, emb_w2, emb_w3,
                                        Z, ZH, MEAN, STDV);

  const dim3 gD(D_MODEL / BN, M_TOK / BM);   // (4, 441)
  const dim3 gF(DFF_ / BN, M_TOK / BM);      // (16, 441)
  const dim3 gH((PRED_ + BN - 1) / BN, (N_BC + BM - 1) / BM);  // (3, 6)

  for (int l = 0; l < 2; ++l) {
    const _Float16* wqh = WQH + (size_t)l * D_MODEL * D_MODEL;
    const _Float16* wkh = WKH + (size_t)l * D_MODEL * D_MODEL;
    const _Float16* wvh = WVH + (size_t)l * D_MODEL * D_MODEL;
    const _Float16* woh = WOH + (size_t)l * D_MODEL * D_MODEL;
    const _Float16* f1h = FF1H + (size_t)l * DFF_ * D_MODEL;
    const _Float16* f2h = FF2H + (size_t)l * D_MODEL * DFF_;

    wmma_gemm_kernel<true,  false><<<gD, 128, 0, stream>>>(ZH, wqh, bq + l * D_MODEL, Qb, M_TOK, D_MODEL, D_MODEL);
    wmma_gemm_kernel<true,  false><<<gD, 128, 0, stream>>>(ZH, wkh, bk + l * D_MODEL, Kb, M_TOK, D_MODEL, D_MODEL);
    wmma_gemm_kernel<true,  false><<<gD, 128, 0, stream>>>(ZH, wvh, bv + l * D_MODEL, Vb, M_TOK, D_MODEL, D_MODEL);
    attention_kernel<<<dim3(N_BC, 8), 192, 0, stream>>>(Qb, Kb, Vb, ATTN);
    wmma_gemm_kernel<false, false><<<gD, 128, 0, stream>>>(ATTN, woh, bo + l * D_MODEL, TMP, M_TOK, D_MODEL, D_MODEL);
    add_ln_kernel<<<M_TOK, 256, 0, stream>>>(Z, TMP, ln1_g + l * D_MODEL, ln1_b + l * D_MODEL, Z, ZH);
    wmma_gemm_kernel<true,  true ><<<gF, 128, 0, stream>>>(ZH, f1h, ff1_b + l * DFF_, HIDD, M_TOK, DFF_, D_MODEL);
    wmma_gemm_kernel<false, false><<<gD, 128, 0, stream>>>(HIDD, f2h, ff2_b + l * D_MODEL, TMP, M_TOK, D_MODEL, DFF_);
    add_ln_kernel<<<M_TOK, 256, 0, stream>>>(Z, TMP, ln2_g + l * D_MODEL, ln2_b + l * D_MODEL, Z, ZH);
  }

  bn_head_kernel<<<4096, 256, 0, stream>>>(Z, bn_g, bn_b, HIN);
  wmma_gemm_kernel<false, false><<<gH, 128, 0, stream>>>(HIN, HWH, head_b, HOUT, N_BC, PRED_, K_HEAD);
  finalize_kernel<<<(BATCH_ * PRED_ * CVAR_ + 255) / 256, 256, 0, stream>>>(HOUT, MEAN, STDV, out);
}
